// AttentionBlock_56891136803184
// MI455X (gfx1250) — matrix operand order repaired and checked by reading, not compiled, not run on hardware
//
#include <hip/hip_runtime.h>

// CDNA5 / gfx1250 attention block, WMMA f16 (16x16x32) with f32 accumulation.
// Pipeline: [K0] w->f16  [K1] groupnorm(+transpose)  [K2] qkv gemm (32x32/wave,
// double-buffered)  [K3] flash attention (pipelined K tiles)  [K4] proj gemm +
// bias + residual (32x32/wave, double-buffered)

typedef __attribute__((ext_vector_type(16))) _Float16 v16h;
typedef __attribute__((ext_vector_type(8)))  _Float16 v8h;
typedef __attribute__((ext_vector_type(8)))  float    v8f;

#define B_      8
#define C_      512
#define T_      1024
#define HEADS_  8
#define CH_     64
#define THREEC_ 1536

__device__ __forceinline__ v8f wmma32(v16h a, v16h b, v8f c) {
  // (neg_a, A, neg_b, B, c_mod, C, reuse_a, reuse_b)
  return __builtin_amdgcn_wmma_f32_16x16x32_f16(false, a, false, b, (short)0, c,
                                                false, false);
}

// Build 16-bit A fragment from two contiguous 8-half chunks (lane-half K split).
__device__ __forceinline__ v16h packA(v8h lo, v8h hi) {
  v16h a;
#pragma unroll
  for (int i = 0; i < 8; ++i) { a[i] = lo[i]; a[i + 8] = hi[i]; }
  return a;
}

// A fragment: 16 rows (lane&15), K-slice split {0..15}/{16..31}.
__device__ __forceinline__ v16h ldA(const _Float16* row, int k, int base8) {
  return packA(*(const v8h*)(row + k + 2 * base8),
               *(const v8h*)(row + k + 2 * base8 + 8));
}

// ---------------- K0: convert weights to f16 ----------------
__global__ void k0_convert(const float* __restrict__ qkv_w,
                           const float* __restrict__ proj_w,
                           _Float16* __restrict__ qw16,
                           _Float16* __restrict__ pw16) {
  int i = blockIdx.x * 256 + threadIdx.x;
  if (i < THREEC_ * C_) qw16[i] = (_Float16)qkv_w[i];
  if (i < C_ * C_)      pw16[i] = (_Float16)proj_w[i];
}

// ---------------- K1: groupnorm, write xn transposed [b][t][c] (f16) --------
__global__ void k1_groupnorm(const float* __restrict__ x,
                             const float* __restrict__ nw,
                             const float* __restrict__ nb,
                             _Float16* __restrict__ xn_t) {
  const int b = blockIdx.x >> 5;
  const int g = blockIdx.x & 31;
  const float* xp = x + ((size_t)b * C_ + g * 16) * T_;  // 16 ch x 1024 t

  float s = 0.f, ss = 0.f;
  for (int i = threadIdx.x; i < 16 * T_; i += 256) {
    float v = xp[i]; s += v; ss += v * v;
  }
  __shared__ float sh1[256], sh2[256];
  sh1[threadIdx.x] = s; sh2[threadIdx.x] = ss;
  __syncthreads();
  for (int off = 128; off > 0; off >>= 1) {
    if (threadIdx.x < off) {
      sh1[threadIdx.x] += sh1[threadIdx.x + off];
      sh2[threadIdx.x] += sh2[threadIdx.x + off];
    }
    __syncthreads();
  }
  const float inv_n = 1.0f / (16.0f * T_);
  const float mu  = sh1[0] * inv_n;
  const float var = sh2[0] * inv_n - mu * mu;
  const float rstd = rsqrtf(var + 1e-5f);

  for (int i = threadIdx.x; i < 16 * T_; i += 256) {
    const int c = g * 16 + (i >> 10);
    const int t = i & (T_ - 1);
    const float v = (xp[i] - mu) * rstd * nw[c] + nb[c];
    xn_t[((size_t)b * T_ + t) * C_ + c] = (_Float16)v;
  }
}

// ---------------- K2: qkv gemm. out[o,t] = sum_c W[o,c]*xn[c,t] + b[o] ------
// Wave computes a 32(o) x 32(t) tile with double-buffered fragments.
// q->[bh][t][ch]*0.125, k->[bh][t][ch], v->[bh][ch][t].
__global__ void k2_qkv(const _Float16* __restrict__ qw16,
                       const _Float16* __restrict__ xn_t,
                       const float* __restrict__ qkv_b,
                       _Float16* __restrict__ q_t,
                       _Float16* __restrict__ k_t,
                       _Float16* __restrict__ vbuf) {
  const int lane = threadIdx.x & 31;
  const int wave = threadIdx.x >> 5;
  const int hi   = lane >> 4;
  const int ln   = lane & 15;
  const int base8 = hi ? 8 : 0;
  const int koff  = hi ? 16 : 0;

  const int wi  = blockIdx.x * 8 + wave;       // 0..12287
  const int b   = wi / (48 * 32);
  const int rem = wi % (48 * 32);
  const int o0  = (rem / 32) * 32;
  const int t0  = (rem % 32) * 32;

  const _Float16* arow0 = qw16 + (size_t)(o0 + ln) * C_;
  const _Float16* arow1 = arow0 + 16 * C_;
  const _Float16* brow0 = xn_t + ((size_t)b * T_ + t0 + ln) * C_;
  const _Float16* brow1 = brow0 + 16 * C_;

  v8f acc00 = {}, acc01 = {}, acc10 = {}, acc11 = {};
  v16h A0 = ldA(arow0, 0, base8), A1 = ldA(arow1, 0, base8);
  v16h B0 = *(const v16h*)(brow0 + koff), B1 = *(const v16h*)(brow1 + koff);
#pragma unroll 4
  for (int k = 0; k < C_; k += 32) {
    const int kn = (k + 32) & (C_ - 1);  // wrap: pipelined tail stays in-bounds
    v16h A0n = ldA(arow0, kn, base8), A1n = ldA(arow1, kn, base8);
    v16h B0n = *(const v16h*)(brow0 + kn + koff);
    v16h B1n = *(const v16h*)(brow1 + kn + koff);
    acc00 = wmma32(A0, B0, acc00);
    acc01 = wmma32(A0, B1, acc01);
    acc10 = wmma32(A1, B0, acc10);
    acc11 = wmma32(A1, B1, acc11);
    A0 = A0n; A1 = A1n; B0 = B0n; B1 = B1n;
  }

  v8f accs[2][2] = {{acc00, acc01}, {acc10, acc11}};
#pragma unroll
  for (int mi = 0; mi < 2; ++mi) {
#pragma unroll
    for (int r = 0; r < 8; ++r) {
      const int o = o0 + mi * 16 + r + 8 * hi;
      const float bs = qkv_b[o];
      if (o < C_) {                                  // q (scaled by ch^-0.5)
        const int h = o >> 6, c = o & 63;
        _Float16* dst = q_t + (size_t)(b * HEADS_ + h) * T_ * CH_ + c;
#pragma unroll
        for (int nj = 0; nj < 2; ++nj)
          dst[(size_t)(t0 + nj * 16 + ln) * CH_] =
              (_Float16)((accs[mi][nj][r] + bs) * 0.125f);
      } else if (o < 2 * C_) {                       // k
        const int oo = o - C_, h = oo >> 6, c = oo & 63;
        _Float16* dst = k_t + (size_t)(b * HEADS_ + h) * T_ * CH_ + c;
#pragma unroll
        for (int nj = 0; nj < 2; ++nj)
          dst[(size_t)(t0 + nj * 16 + ln) * CH_] =
              (_Float16)(accs[mi][nj][r] + bs);
      } else {                                       // v  [bh][ch][t]
        const int oo = o - 2 * C_, h = oo >> 6, c = oo & 63;
        _Float16* dst = vbuf + ((size_t)(b * HEADS_ + h) * CH_ + c) * T_;
#pragma unroll
        for (int nj = 0; nj < 2; ++nj)
          dst[t0 + nj * 16 + ln] = (_Float16)(accs[mi][nj][r] + bs);
      }
    }
  }
}

// ---------------- K3: flash attention per wave (16 queries) -----------------
__global__ void k3_attn(const _Float16* __restrict__ q_t,
                        const _Float16* __restrict__ k_t,
                        const _Float16* __restrict__ vbuf,
                        _Float16* __restrict__ a_t) {
  __shared__ __align__(32) _Float16 ldsP[8][16][32];  // per-wave P staging
  const int lane = threadIdx.x & 31;
  const int wave = threadIdx.x >> 5;
  const int hi   = lane >> 4;
  const int ln   = lane & 15;
  const int base8 = hi ? 8 : 0;
  const int koff  = hi ? 16 : 0;

  const int bh = blockIdx.x >> 3;                       // 0..63
  const int t0 = (((blockIdx.x & 7) << 3) + wave) * 16; // query tile
  const int b = bh >> 3, h = bh & 7;

  // Q as A fragments (K = ch, two 32-wide slices)
  const _Float16* qrow = q_t + ((size_t)bh * T_ + t0 + ln) * CH_;
  const v16h Aq0 = ldA(qrow, 0, base8);
  const v16h Aq1 = ldA(qrow, 32, base8);

  float mrow[8], lrow[8];
  v8f zero = {};
  v8f O[4] = {zero, zero, zero, zero};
#pragma unroll
  for (int r = 0; r < 8; ++r) { mrow[r] = -1e30f; lrow[r] = 0.f; }

  // double-buffered K fragments: [s-subtile][ch-slice]
  const _Float16* kbp = k_t + (size_t)bh * T_ * CH_ + (size_t)ln * CH_;
  v16h Bk00 = *(const v16h*)(kbp + koff);
  v16h Bk01 = *(const v16h*)(kbp + 32 + koff);
  v16h Bk10 = *(const v16h*)(kbp + 16 * CH_ + koff);
  v16h Bk11 = *(const v16h*)(kbp + 16 * CH_ + 32 + koff);

  for (int s0 = 0; s0 < T_; s0 += 32) {
    v8f S0 = zero, S1 = zero;
    S0 = wmma32(Aq0, Bk00, S0);
    S0 = wmma32(Aq1, Bk01, S0);
    S1 = wmma32(Aq0, Bk10, S1);
    S1 = wmma32(Aq1, Bk11, S1);

    // prefetch next s-block's K tiles (wrap keeps addresses in-bounds)
    const int sn = (s0 + 32) & (T_ - 1);
    const _Float16* kpn = kbp + (size_t)sn * CH_;
    Bk00 = *(const v16h*)(kpn + koff);
    Bk01 = *(const v16h*)(kpn + 32 + koff);
    Bk10 = *(const v16h*)(kpn + 16 * CH_ + koff);
    Bk11 = *(const v16h*)(kpn + 16 * CH_ + 32 + koff);

    float alpha[8];
#pragma unroll
    for (int r = 0; r < 8; ++r) {
      float rm = fmaxf(S0[r], S1[r]);
#pragma unroll
      for (int msk = 1; msk < 16; msk <<= 1)
        rm = fmaxf(rm, __shfl_xor(rm, msk, 32));
      const float mo = mrow[r];
      const float mn = fmaxf(mo, rm);
      mrow[r]  = mn;
      alpha[r] = __expf(mo - mn);
      const float p0 = __expf(S0[r] - mn);
      const float p1 = __expf(S1[r] - mn);
      float rs = p0 + p1;
#pragma unroll
      for (int msk = 1; msk < 16; msk <<= 1)
        rs += __shfl_xor(rs, msk, 32);
      lrow[r] = lrow[r] * alpha[r] + rs;
      // D layout -> A layout via LDS (per-wave private tile)
      ldsP[wave][r + 8 * hi][ln]      = (_Float16)p0;
      ldsP[wave][r + 8 * hi][16 + ln] = (_Float16)p1;
    }
    asm volatile("s_wait_dscnt 0" ::: "memory");
    const v16h Ap = packA(*(const v8h*)&ldsP[wave][ln][2 * base8],
                          *(const v8h*)&ldsP[wave][ln][2 * base8 + 8]);
#pragma unroll
    for (int j = 0; j < 4; ++j) {
#pragma unroll
      for (int r = 0; r < 8; ++r) O[j][r] *= alpha[r];
      const _Float16* vp =
          vbuf + ((size_t)bh * CH_ + j * 16 + ln) * T_ + s0 + koff;
      O[j] = wmma32(Ap, *(const v16h*)vp, O[j]);
    }
  }

#pragma unroll
  for (int r = 0; r < 8; ++r) lrow[r] = 1.0f / lrow[r];
#pragma unroll
  for (int j = 0; j < 4; ++j)
#pragma unroll
    for (int r = 0; r < 8; ++r) {
      const int t = t0 + r + 8 * hi;
      const int c = h * CH_ + j * 16 + ln;
      a_t[((size_t)b * T_ + t) * C_ + c] = (_Float16)(O[j][r] * lrow[r]);
    }
}

// ---------------- K4: proj gemm + bias + residual ---------------------------
__global__ void k4_proj(const _Float16* __restrict__ pw16,
                        const _Float16* __restrict__ a_t,
                        const float* __restrict__ proj_b,
                        const float* __restrict__ x,
                        float* __restrict__ out) {
  const int lane = threadIdx.x & 31;
  const int wave = threadIdx.x >> 5;
  const int hi   = lane >> 4;
  const int ln   = lane & 15;
  const int base8 = hi ? 8 : 0;
  const int koff  = hi ? 16 : 0;

  const int wi  = blockIdx.x * 8 + wave;   // 0..4095
  const int b   = wi / (16 * 32);
  const int rem = wi % (16 * 32);
  const int o0  = (rem / 32) * 32;
  const int t0  = (rem % 32) * 32;

  const _Float16* arow0 = pw16 + (size_t)(o0 + ln) * C_;
  const _Float16* arow1 = arow0 + 16 * C_;
  const _Float16* brow0 = a_t + ((size_t)b * T_ + t0 + ln) * C_;
  const _Float16* brow1 = brow0 + 16 * C_;

  v8f acc00 = {}, acc01 = {}, acc10 = {}, acc11 = {};
  v16h A0 = ldA(arow0, 0, base8), A1 = ldA(arow1, 0, base8);
  v16h B0 = *(const v16h*)(brow0 + koff), B1 = *(const v16h*)(brow1 + koff);
#pragma unroll 4
  for (int k = 0; k < C_; k += 32) {
    const int kn = (k + 32) & (C_ - 1);
    v16h A0n = ldA(arow0, kn, base8), A1n = ldA(arow1, kn, base8);
    v16h B0n = *(const v16h*)(brow0 + kn + koff);
    v16h B1n = *(const v16h*)(brow1 + kn + koff);
    acc00 = wmma32(A0, B0, acc00);
    acc01 = wmma32(A0, B1, acc01);
    acc10 = wmma32(A1, B0, acc10);
    acc11 = wmma32(A1, B1, acc11);
    A0 = A0n; A1 = A1n; B0 = B0n; B1 = B1n;
  }

  v8f accs[2][2] = {{acc00, acc01}, {acc10, acc11}};
#pragma unroll
  for (int mi = 0; mi < 2; ++mi) {
#pragma unroll
    for (int r = 0; r < 8; ++r) {
      const int o = o0 + mi * 16 + r + 8 * hi;
      const float bs = proj_b[o];
#pragma unroll
      for (int nj = 0; nj < 2; ++nj) {
        const size_t idx = ((size_t)b * C_ + o) * T_ + t0 + nj * 16 + ln;
        out[idx] = x[idx] + bs + accs[mi][nj][r];
      }
    }
  }
}

// ---------------- launch ----------------------------------------------------
extern "C" void kernel_launch(void* const* d_in, const int* in_sizes, int n_in,
                              void* d_out, int out_size, void* d_ws,
                              size_t ws_size, hipStream_t stream) {
  (void)in_sizes; (void)n_in; (void)out_size; (void)ws_size;
  const float* x      = (const float*)d_in[0];
  const float* norm_w = (const float*)d_in[1];
  const float* norm_b = (const float*)d_in[2];
  const float* qkv_w  = (const float*)d_in[3];
  const float* qkv_b  = (const float*)d_in[4];
  const float* proj_w = (const float*)d_in[5];
  const float* proj_b = (const float*)d_in[6];
  float* out = (float*)d_out;

  char* ws = (char*)d_ws;
  _Float16* xn_t = (_Float16*)(ws);                      //  8 MB [b][t][c]
  _Float16* qw16 = (_Float16*)(ws + 8388608);            //  1.5 MB
  _Float16* pw16 = (_Float16*)(ws + 9961472);            //  0.5 MB
  _Float16* q_t  = (_Float16*)(ws + 10485760);           //  8 MB [bh][t][ch]
  _Float16* k_t  = (_Float16*)(ws + 18874368);           //  8 MB [bh][t][ch]
  _Float16* vbuf = (_Float16*)(ws + 27262976);           //  8 MB [bh][ch][t]
  _Float16* a_t  = (_Float16*)(ws + 35651584);           //  8 MB [b][t][c]

  k0_convert <<<3072, 256, 0, stream>>>(qkv_w, proj_w, qw16, pw16);
  k1_groupnorm<<< 256, 256, 0, stream>>>(x, norm_w, norm_b, xn_t);
  k2_qkv     <<<1536, 256, 0, stream>>>(qw16, xn_t, qkv_b, q_t, k_t, vbuf);
  k3_attn    <<< 512, 256, 0, stream>>>(q_t, k_t, vbuf, a_t);
  k4_proj    <<< 512, 256, 0, stream>>>(pw16, a_t, proj_b, x, out);
}
